// GCN_23888608100578
// MI455X (gfx1250) — matrix operand order repaired and checked by reading, not compiled, not run on hardware
//
#include <hip/hip_runtime.h>
#include <hip/hip_bf16.h>

typedef __attribute__((ext_vector_type(16))) _Float16 v16h;
typedef __attribute__((ext_vector_type(8)))  float    v8f;

// ---------------- utility kernels ----------------

__global__ void fill_f32(float* __restrict__ p, float v, long long n) {
    long long i = (long long)blockIdx.x * blockDim.x + threadIdx.x;
    if (i < n) p[i] = v;
}

__global__ void cast_f16(const float* __restrict__ x, _Float16* __restrict__ y, long long n) {
    long long i = (long long)blockIdx.x * blockDim.x + threadIdx.x;
    if (i < n) y[i] = (_Float16)x[i];
}

// W [K][NC] row-major f32 -> WT [NC][K] f16  (so WMMA B operand reads contiguous K pairs)
__global__ void transpose_to_f16(const float* __restrict__ W, _Float16* __restrict__ WT,
                                 int K, int NC) {
    int idx = blockIdx.x * blockDim.x + threadIdx.x;
    if (idx >= K * NC) return;
    int k = idx / NC, n = idx % NC;
    WT[(size_t)n * K + k] = (_Float16)W[idx];
}

// ---------------- graph degree / norm ----------------

__global__ void deg_edges(const long long* __restrict__ dst, float* __restrict__ deg,
                          long long E) {
    long long e = (long long)blockIdx.x * blockDim.x + threadIdx.x;
    if (e < E) atomicAdd(&deg[(int)dst[e]], 1.0f);
}

__global__ void rsqrt_k(const float* __restrict__ deg, float* __restrict__ dinv, int n) {
    int i = blockIdx.x * blockDim.x + threadIdx.x;
    if (i < n) dinv[i] = rsqrtf(deg[i]);   // deg >= 1 (self loop)
}

// ---------------- WMMA GEMM:  C[M][128] = A[M][K]_f16 * B[K][128]  (B given as BT[128][K]) ----
// One wave computes a 16(M) x 128(N) strip with 8 f32 accumulators.
__global__ void gemm_wmma_n128(const _Float16* __restrict__ A, int lda,
                               const _Float16* __restrict__ BT,   // [128][K] f16
                               float* __restrict__ C, int ldc, int coff,
                               int M, int K) {
    int wid  = (blockIdx.x * blockDim.x + threadIdx.x) >> 5;   // global wave id
    int lane = threadIdx.x & 31;
    int m0 = wid * 16;
    if (m0 >= M) return;                    // wave-uniform: EXEC stays all-ones

    int mrow = lane & 15;
    int khi  = (lane >> 4) & 1;

    v8f acc[8];
#pragma unroll
    for (int nt = 0; nt < 8; ++nt)
#pragma unroll
        for (int i = 0; i < 8; ++i) acc[nt][i] = 0.0f;

    for (int ks = 0; ks < K; ks += 32) {
        // A tile 16x32 (ISA 7.12.2 16-bit A layout)
        v16h a;
        const _Float16* Arow = A + (size_t)(m0 + mrow) * lda + ks;
#pragma unroll
        for (int v = 0; v < 8; ++v) {
            int k0 = ((v & 3) << 1) + (khi << 3) + ((v >> 2) << 4);
            a[2 * v]     = Arow[k0];
            a[2 * v + 1] = Arow[k0 + 1];
        }
#pragma unroll
        for (int nt = 0; nt < 8; ++nt) {
            // B tile 32x16: lane = n, element pairs index K (same K layout as A: lanes 16-31 -> K+8)
            v16h b;
            const _Float16* Brow = BT + (size_t)(nt * 16 + (lane & 15)) * K + ks;
#pragma unroll
            for (int v = 0; v < 8; ++v) {
                int k0 = ((v & 3) << 1) + (khi << 3) + ((v >> 2) << 4);
                b[2 * v]     = Brow[k0];
                b[2 * v + 1] = Brow[k0 + 1];
            }
            acc[nt] = __builtin_amdgcn_wmma_f32_16x16x32_f16(
                false, a, false, b, (short)0, acc[nt], false, false);
        }
    }
    // C 16x16 f32 layout: VGPR v -> row m0 + v + 8*khi, lane&15 -> col
#pragma unroll
    for (int nt = 0; nt < 8; ++nt) {
        int col = coff + nt * 16 + (lane & 15);
#pragma unroll
        for (int v = 0; v < 8; ++v) {
            int row = m0 + v + (khi << 3);
            C[(size_t)row * ldc + col] = acc[nt][v];
        }
    }
}

// ---------------- GCN propagation ----------------

// hcat[n, coff..coff+127] = dinv[n]^2 * t[n,:]   (self-loop term, also initializes slice)
__global__ void selfloop_init(const float* __restrict__ t, const float* __restrict__ dinv,
                              float* __restrict__ hcat, int ldh, int coff, int n) {
    int idx = blockIdx.x * blockDim.x + threadIdx.x;
    if (idx >= n * 128) return;
    int node = idx >> 7, c = idx & 127;
    float di = dinv[node];
    hcat[(size_t)node * ldh + coff + c] = di * di * t[idx];
}

// one wave per edge, 4 features per lane, scatter-add (L2-resident targets)
__global__ void edge_prop(const long long* __restrict__ src, const long long* __restrict__ dst,
                          const float* __restrict__ dinv, const float* __restrict__ t,
                          float* __restrict__ hcat, int ldh, int coff, long long E) {
    long long e = (long long)blockIdx.x * (blockDim.x >> 5) + (threadIdx.x >> 5);
    if (e >= E) return;
    int s = (int)src[e], d = (int)dst[e];
    float w = dinv[s] * dinv[d];
    int lane = threadIdx.x & 31;
    const float* ts = t + (size_t)s * 128;
    float* od = hcat + (size_t)d * ldh + coff;
#pragma unroll
    for (int i = 0; i < 4; ++i) {
        int c = lane + i * 32;
        atomicAdd(od + c, w * ts[c]);
    }
}

// ---------------- BatchNorm (training-mode, biased stats) ----------------

__global__ void bn_stats(const float* __restrict__ x, int rows, int ld, int coff,
                         float* __restrict__ sums, float* __restrict__ sumsq) {
    int c = threadIdx.x;                     // blockDim.x == 128
    float s = 0.f, s2 = 0.f;
    for (int r = blockIdx.x; r < rows; r += gridDim.x) {
        float v = x[(size_t)r * ld + coff + c];
        s += v; s2 += v * v;
    }
    atomicAdd(&sums[c], s);
    atomicAdd(&sumsq[c], s2);
}

__global__ void bn_coeffs(const float* __restrict__ sums, const float* __restrict__ sumsq,
                          const float* __restrict__ g, const float* __restrict__ bt,
                          float rows, float* __restrict__ a, float* __restrict__ b) {
    int c = threadIdx.x;                     // 128 threads
    float mu  = sums[c] / rows;
    float var = sumsq[c] / rows - mu * mu;
    float sc  = g[c] * rsqrtf(var + 1e-5f);
    a[c] = sc;
    b[c] = bt[c] - mu * sc;
}

__global__ void bn_apply_relu(float* __restrict__ x, int rows, int ld, int coff,
                              const float* __restrict__ a, const float* __restrict__ b,
                              _Float16* __restrict__ h16 /* nullable */) {
    int idx = blockIdx.x * blockDim.x + threadIdx.x;
    if (idx >= rows * 128) return;
    int r = idx >> 7, c = idx & 127;
    size_t off = (size_t)r * ld + coff + c;
    float v = fmaxf(x[off] * a[c] + b[c], 0.0f);
    x[off] = v;
    if (h16) h16[idx] = (_Float16)v;
}

// ---------------- pooling ----------------

__global__ void pool_accum(const float* __restrict__ hcat, const long long* __restrict__ batch,
                           float* __restrict__ psum, float* __restrict__ cnt, int n) {
    int idx = blockIdx.x * blockDim.x + threadIdx.x;
    if (idx >= n * 384) return;
    int node = idx / 384, c = idx % 384;
    int g = (int)batch[node];
    atomicAdd(&psum[(size_t)g * 384 + c], hcat[(size_t)node * 384 + c]);
    if (c == 0) atomicAdd(&cnt[g], 1.0f);
}

__global__ void pool_finalize(const float* __restrict__ psum, const float* __restrict__ cnt,
                              _Float16* __restrict__ p16, int total) {
    int idx = blockIdx.x * blockDim.x + threadIdx.x;
    if (idx >= total) return;
    int g = idx / 384;
    float c = fmaxf(cnt[g], 1.0f);
    p16[idx] = (_Float16)(psum[idx] / c);
}

// ---------------- head: lin2 + log_softmax ----------------

__global__ void lin2_logsoftmax(const float* __restrict__ z, const float* __restrict__ W2,
                                const float* __restrict__ b2, float* __restrict__ out, int G) {
    int r = blockIdx.x * blockDim.x + threadIdx.x;
    if (r >= G) return;
    float logits[10];
#pragma unroll
    for (int j = 0; j < 10; ++j) logits[j] = b2[j];
    for (int k = 0; k < 128; ++k) {
        float zv = z[(size_t)r * 128 + k];
#pragma unroll
        for (int j = 0; j < 10; ++j) logits[j] += zv * W2[k * 10 + j];
    }
    float mx = logits[0];
#pragma unroll
    for (int j = 1; j < 10; ++j) mx = fmaxf(mx, logits[j]);
    float se = 0.f;
#pragma unroll
    for (int j = 0; j < 10; ++j) se += __expf(logits[j] - mx);
    float lse = mx + __logf(se);
#pragma unroll
    for (int j = 0; j < 10; ++j) out[(size_t)r * 10 + j] = logits[j] - lse;
}

// ---------------- host driver ----------------

static inline unsigned grd(long long n, int b) { return (unsigned)((n + b - 1) / b); }

extern "C" void kernel_launch(void* const* d_in, const int* in_sizes, int n_in,
                              void* d_out, int out_size, void* d_ws, size_t ws_size,
                              hipStream_t stream) {
    const float*     x     = (const float*)d_in[0];
    const long long* ei    = (const long long*)d_in[1];
    const long long* batch = (const long long*)d_in[2];
    const float*     W1    = (const float*)d_in[3];
    const float*     g1    = (const float*)d_in[5];
    const float*     bt1   = (const float*)d_in[6];
    const float*     Wc    = (const float*)d_in[7];
    const float*     gc    = (const float*)d_in[9];
    const float*     btc   = (const float*)d_in[10];
    const float*     Wl1   = (const float*)d_in[11];
    const float*     gl    = (const float*)d_in[13];
    const float*     btl   = (const float*)d_in[14];
    const float*     Wl2   = (const float*)d_in[15];
    const float*     bl2   = (const float*)d_in[16];
    float*           out   = (float*)d_out;

    const int       N = in_sizes[0] / 128;
    const long long E = in_sizes[1] / 2;
    const int       G = out_size / 10;
    const long long* src = ei;
    const long long* dst = ei + E;

    // workspace carve (all re-initialized every call)
    char* p = (char*)d_ws;
    auto alloc = [&](size_t bytes) -> void* {
        void* r = (void*)p; p += (bytes + 255) & ~(size_t)255; return r;
    };
    float*    hcat     = (float*)alloc((size_t)N * 384 * 4);
    float*    t        = (float*)alloc((size_t)N * 128 * 4);
    _Float16* h16      = (_Float16*)alloc((size_t)N * 128 * 2);
    float*    deg      = (float*)alloc((size_t)N * 4);
    float*    dinv     = (float*)alloc((size_t)N * 4);
    _Float16* WT16     = (_Float16*)alloc((size_t)128 * 384 * 2);
    float*    bn_sums  = (float*)alloc(128 * 4);
    float*    bn_sumsq = (float*)alloc(128 * 4);
    float*    bn_a     = (float*)alloc(128 * 4);
    float*    bn_b     = (float*)alloc(128 * 4);
    float*    psum     = (float*)alloc((size_t)G * 384 * 4);
    float*    cnt      = (float*)alloc((size_t)G * 4);
    _Float16* pooled16 = (_Float16*)alloc((size_t)G * 384 * 2);
    float*    z        = (float*)alloc((size_t)G * 128 * 4);

    const int B = 256;

    // degrees with self-loop, dinv = deg^-1/2
    fill_f32<<<grd(N, B), B, 0, stream>>>(deg, 1.0f, N);
    deg_edges<<<grd(E, B), B, 0, stream>>>(dst, deg, E);
    rsqrt_k<<<grd(N, B), B, 0, stream>>>(deg, dinv, N);

    // x -> f16 GEMM operand
    cast_f16<<<grd((long long)N * 128, B), B, 0, stream>>>(x, h16, (long long)N * 128);

    // three GCN layers (BN makes conv biases no-ops, skipped)
    for (int layer = 0; layer < 3; ++layer) {
        const float* W  = (layer == 0) ? W1  : Wc  + (size_t)(layer - 1) * 128 * 128;
        const float* g  = (layer == 0) ? g1  : gc  + (size_t)(layer - 1) * 128;
        const float* bt = (layer == 0) ? bt1 : btc + (size_t)(layer - 1) * 128;
        const int coff = layer * 128;

        transpose_to_f16<<<grd(128 * 128, B), B, 0, stream>>>(W, WT16, 128, 128);
        // t = h @ W  via WMMA
        gemm_wmma_n128<<<grd((long long)(N / 16) * 32, B), B, 0, stream>>>(
            h16, 128, WT16, t, 128, 0, N, 128);
        // propagate: self-loop init then edge scatter-add into hcat slice
        selfloop_init<<<grd((long long)N * 128, B), B, 0, stream>>>(t, dinv, hcat, 384, coff, N);
        edge_prop<<<grd(E * 32, B), B, 0, stream>>>(src, dst, dinv, t, hcat, 384, coff, E);
        // BN + ReLU (in place) and recast to f16 for next GEMM
        fill_f32<<<1, 128, 0, stream>>>(bn_sums, 0.0f, 128);
        fill_f32<<<1, 128, 0, stream>>>(bn_sumsq, 0.0f, 128);
        bn_stats<<<512, 128, 0, stream>>>(hcat, N, 384, coff, bn_sums, bn_sumsq);
        bn_coeffs<<<1, 128, 0, stream>>>(bn_sums, bn_sumsq, g, bt, (float)N, bn_a, bn_b);
        bn_apply_relu<<<grd((long long)N * 128, B), B, 0, stream>>>(
            hcat, N, 384, coff, bn_a, bn_b, h16);
    }

    // global mean pool over graphs
    fill_f32<<<grd((long long)G * 384, B), B, 0, stream>>>(psum, 0.0f, (long long)G * 384);
    fill_f32<<<grd(G, B), B, 0, stream>>>(cnt, 0.0f, G);
    pool_accum<<<grd((long long)N * 384, B), B, 0, stream>>>(hcat, batch, psum, cnt, N);
    pool_finalize<<<grd((long long)G * 384, B), B, 0, stream>>>(psum, cnt, pooled16, G * 384);

    // lin1 (WMMA, K=384) + BN + ReLU
    transpose_to_f16<<<grd(384 * 128, B), B, 0, stream>>>(Wl1, WT16, 384, 128);
    gemm_wmma_n128<<<grd((long long)(G / 16) * 32, B), B, 0, stream>>>(
        pooled16, 384, WT16, z, 128, 0, G, 384);
    fill_f32<<<1, 128, 0, stream>>>(bn_sums, 0.0f, 128);
    fill_f32<<<1, 128, 0, stream>>>(bn_sumsq, 0.0f, 128);
    bn_stats<<<512, 128, 0, stream>>>(z, G, 128, 0, bn_sums, bn_sumsq);
    bn_coeffs<<<1, 128, 0, stream>>>(bn_sums, bn_sumsq, gl, btl, (float)G, bn_a, bn_b);
    bn_apply_relu<<<grd((long long)G * 128, B), B, 0, stream>>>(
        z, G, 128, 0, bn_a, bn_b, (_Float16*)nullptr);

    // lin2 + log_softmax
    lin2_logsoftmax<<<grd(G, B), B, 0, stream>>>(z, Wl2, bl2, out, G);
}
